// SAGE_5385888989904
// MI455X (gfx1250) — matrix operand order repaired and checked by reading, not compiled, not run on hardware
//
#include <hip/hip_runtime.h>
#include <hip/hip_bf16.h>

// ---------------------------------------------------------------------------
// GraphSAGE 2-layer forward on gfx1250 (MI455X).
//   GEMMs: v_wmma_f32_16x16x32_bf16, A staged in LDS as bf16 (2x ds_load_b128
//   per fragment), B pre-swizzled in workspace to fragment-linear order
//   (2x coalesced global_load_b128 per fragment). 80 rows per block amortize
//   weight traffic 5x. Scatter/aggregate uses global_atomic_add_f32.
// ---------------------------------------------------------------------------

typedef __attribute__((ext_vector_type(16))) __bf16 v16bf;
typedef __attribute__((ext_vector_type(8)))  float  v8f;

constexpr int Nn = 50000;
constexpr int Ee = 800000;
constexpr int NF = 128;
constexpr int NH = 256;

constexpr int ROWS = 80;          // rows per GEMM block (50000 = 625 * 80)
constexpr int RT   = ROWS / 16;   // row tiles per block

// workspace layout (float element offsets)
constexpr size_t OFF_DEG   = 0;
constexpr size_t OFF_BNSUM = OFF_DEG + Nn;
constexpr size_t OFF_BNSQ  = OFF_BNSUM + NH;
constexpr size_t OFF_AGG1  = OFF_BNSQ + NH;
constexpr size_t OFF_AGG2  = OFF_AGG1 + (size_t)Nn * NF;
constexpr size_t ZERO_TOT  = OFF_AGG2 + (size_t)Nn * NH;   // accumulators to zero
constexpr size_t OFF_H     = ZERO_TOT;                     // hidden activations [Nn, NH]
constexpr size_t F_TOT     = OFF_H + (size_t)Nn * NH;      // bf16 weights start here (32B aligned)

// ---------------------------------------------------------------------------
__global__ void k_zero(float* __restrict__ p, size_t n) {
  size_t i = (size_t)blockIdx.x * blockDim.x + threadIdx.x;
  if (i < n) p[i] = 0.0f;
}

// Swizzle f32 weight [KDIM, 256] into WMMA B-fragment-linear bf16:
//   dst[((w*KT + kt)*32 + lane)*16 + e] = W[(kt*32 + (e>>3)*16 + (lane>>4)*8 + (e&7))*256 + w*16 + (lane&15)]
// so a lane's 16 fragment elements are contiguous (one 32B vector load).
template <int KDIM>
__global__ void k_swizzleW(const float* __restrict__ W, __bf16* __restrict__ dst) {
  constexpr int KT = KDIM / 32;
  int i = blockIdx.x * blockDim.x + threadIdx.x;
  if (i >= KDIM * NH) return;
  int e    = i & 15;
  int lane = (i >> 4) & 31;
  int frag = i >> 9;
  int kt   = frag % KT;
  int w    = frag / KT;
  int g = lane >> 4, l = lane & 15;
  int K   = kt * 32 + (e >> 3) * 16 + g * 8 + (e & 7);
  int col = w * 16 + l;
  dst[i] = (__bf16)W[K * NH + col];
}

__global__ void k_degree(const long long* __restrict__ ei, float* __restrict__ deg) {
  int e = blockIdx.x * blockDim.x + threadIdx.x;
  if (e < Ee) atomicAdd(&deg[(int)ei[Ee + e]], 1.0f);
}

// one wave (32 lanes) per edge; each lane owns 4 of the 128 features
__global__ void k_scatter128(const long long* __restrict__ ei, const float* __restrict__ x,
                             float* __restrict__ agg) {
  long long t = (long long)blockIdx.x * blockDim.x + threadIdx.x;
  long long e = t >> 5;
  int lane = (int)(t & 31);
  if (e >= Ee) return;
  long long s = ei[e];
  long long d = ei[Ee + e];
  float4 v = ((const float4*)x)[s * 32 + lane];
  float* dp = agg + d * 128 + lane * 4;
  atomicAdd(dp + 0, v.x); atomicAdd(dp + 1, v.y);
  atomicAdd(dp + 2, v.z); atomicAdd(dp + 3, v.w);
}

// one wave per edge; each lane owns 8 of the 256 features
__global__ void k_scatter256(const long long* __restrict__ ei, const float* __restrict__ h,
                             float* __restrict__ agg) {
  long long t = (long long)blockIdx.x * blockDim.x + threadIdx.x;
  long long e = t >> 5;
  int lane = (int)(t & 31);
  if (e >= Ee) return;
  long long s = ei[e];
  long long d = ei[Ee + e];
  const float4* hb = (const float4*)h;
  float4 v0 = hb[s * 64 + lane * 2 + 0];
  float4 v1 = hb[s * 64 + lane * 2 + 1];
  float* dp = agg + d * 256 + lane * 8;
  atomicAdd(dp + 0, v0.x); atomicAdd(dp + 1, v0.y);
  atomicAdd(dp + 2, v0.z); atomicAdd(dp + 3, v0.w);
  atomicAdd(dp + 4, v1.x); atomicAdd(dp + 5, v1.y);
  atomicAdd(dp + 6, v1.z); atomicAdd(dp + 7, v1.w);
}

__global__ void k_bn_apply(float* __restrict__ h, const float* __restrict__ bnsum,
                           const float* __restrict__ bnsq, const float* __restrict__ gamma,
                           const float* __restrict__ beta) {
  size_t i = (size_t)blockIdx.x * blockDim.x + threadIdx.x;
  if (i >= (size_t)Nn * NH) return;
  int c = (int)(i & (NH - 1));
  const float inv_n = 1.0f / (float)Nn;
  float mu  = bnsum[c] * inv_n;
  float var = bnsq[c] * inv_n - mu * mu;
  h[i] = (h[i] - mu) * rsqrtf(var + 1e-5f) * gamma[c] + beta[c];
}

// ---------------------------------------------------------------------------
__device__ __forceinline__ unsigned pack_bf16(float a, float b) {
  __bf16 lo = (__bf16)a, hi = (__bf16)b;
  return (unsigned)__builtin_bit_cast(unsigned short, lo) |
         ((unsigned)__builtin_bit_cast(unsigned short, hi) << 16);
}

// A fragment (16x32 MxK, bf16): lane l<16 -> M=l ; element i -> K = (i>>3)*16 + g*8 + (i&7)
// From a row-major bf16 LDS tile this is two contiguous 16B chunks -> 2x ds_load_b128.
__device__ __forceinline__ v16bf load_a_frag(const __bf16* At, int ldk, int l, int g, int kb) {
  union { v16bf v; uint4 q[2]; } u;
  const uint4* p = (const uint4*)(At + l * ldk + kb + g * 8);
  u.q[0] = p[0];   // K = kb + g*8 + [0..7]
  u.q[1] = p[2];   // K = kb + 16 + g*8 + [0..7]
  return u.v;
}

// Fused SAGE layer: out = (agg*invdeg)@Wl + act@Wr + bias, row-L2-normalize,
// then (FIRST) relu + BN-stat accumulation, else direct store.
// Block = 512 threads = 16 waves; block covers 80 rows x 256 cols.
// Wave w owns column tile w and iterates 5 row tiles, reusing B fragments.
template <int KDIM, bool FIRST>
__global__ __launch_bounds__(512) void k_sage_gemm(
    const float* __restrict__ agg, const float* __restrict__ act,
    const float* __restrict__ deg,
    const __bf16* __restrict__ Wl, const __bf16* __restrict__ Wr,  // swizzled
    const float* __restrict__ bias, float* __restrict__ out,
    float* __restrict__ bnsum, float* __restrict__ bnsq) {
  constexpr int KT = KDIM / 32;
  __shared__ __align__(16) __bf16 Atile[ROWS * KDIM];   // invdeg-scaled neighbors
  __shared__ __align__(16) __bf16 Xtile[ROWS * KDIM];   // self features
  __shared__ float invdS[ROWS];
  __shared__ float rowsq[ROWS];
  __shared__ float invn[ROWS];

  const int tid = threadIdx.x;
  const int m0  = blockIdx.x * ROWS;
  const int lane = tid & 31;
  const int w    = tid >> 5;          // 16 waves -> 16 column tiles
  const int l    = lane & 15;
  const int g    = lane >> 4;
  const int col0 = w * 16;

  const v16bf* WlV = (const v16bf*)Wl;
  const v16bf* WrV = (const v16bf*)Wr;

  // hint the weight slices for this wave into cache while we stage LDS
#pragma unroll
  for (int kt = 0; kt < KT; ++kt) {
    __builtin_prefetch(&WlV[(w * KT + kt) * 32 + lane], 0, 3);
    __builtin_prefetch(&WrV[(w * KT + kt) * 32 + lane], 0, 3);
  }

  if (tid < ROWS) {
    invdS[tid] = 1.0f / fmaxf(deg[m0 + tid], 1.0f);
    rowsq[tid] = 0.0f;
  }
  __syncthreads();

  // stage A (scaled) and X tiles as packed bf16 pairs
  {
    const float2* aggp = (const float2*)(agg + (size_t)m0 * KDIM);
    const float2* actp = (const float2*)(act + (size_t)m0 * KDIM);
    unsigned* At2 = (unsigned*)Atile;
    unsigned* Xt2 = (unsigned*)Xtile;
    constexpr int PAIRS = ROWS * KDIM / 2;
    for (int p = tid; p < PAIRS; p += 512) {
      int r = p / (KDIM / 2);
      float s = invdS[r];
      float2 va = aggp[p];
      float2 vx = actp[p];
      At2[p] = pack_bf16(va.x * s, va.y * s);
      Xt2[p] = pack_bf16(vx.x, vx.y);
    }
  }
  __syncthreads();

  v8f acc[RT];
#pragma unroll
  for (int rt = 0; rt < RT; ++rt) acc[rt] = (v8f){};

#pragma unroll
  for (int kt = 0; kt < KT; ++kt) {
    const int kb = kt * 32;
    v16bf bl = WlV[(w * KT + kt) * 32 + lane];   // 2x global_load_b128, coalesced
    v16bf br = WrV[(w * KT + kt) * 32 + lane];
#pragma unroll
    for (int rt = 0; rt < RT; ++rt) {
      v16bf aA = load_a_frag(Atile + rt * 16 * KDIM, KDIM, l, g, kb);
      acc[rt] = __builtin_amdgcn_wmma_f32_16x16x32_bf16(false, aA, false, bl,
                                                        (short)0, acc[rt], false, false);
      v16bf aX = load_a_frag(Xtile + rt * 16 * KDIM, KDIM, l, g, kb);
      acc[rt] = __builtin_amdgcn_wmma_f32_16x16x32_bf16(false, aX, false, br,
                                                        (short)0, acc[rt], false, false);
    }
  }

  // epilogue: bias, row L2-norm reduction (16 lanes x 16 waves -> LDS atomics)
  const int col = col0 + l;
  const float bv = bias[col];
#pragma unroll
  for (int rt = 0; rt < RT; ++rt) {
    float sq[8];
#pragma unroll
    for (int r = 0; r < 8; ++r) {
      float v = acc[rt][r] + bv;
      acc[rt][r] = v;
      sq[r] = v * v;
    }
#pragma unroll
    for (int off = 1; off < 16; off <<= 1) {
#pragma unroll
      for (int r = 0; r < 8; ++r) sq[r] += __shfl_xor(sq[r], off, 32);
    }
    if (l == 0) {
#pragma unroll
      for (int r = 0; r < 8; ++r) atomicAdd(&rowsq[rt * 16 + g * 8 + r], sq[r]);
    }
  }
  __syncthreads();
  if (tid < ROWS) invn[tid] = 1.0f / fmaxf(sqrtf(rowsq[tid]), 1e-12f);
  __syncthreads();

  float s1 = 0.0f, s2 = 0.0f;
#pragma unroll
  for (int rt = 0; rt < RT; ++rt) {
#pragma unroll
    for (int r = 0; r < 8; ++r) {
      int row = rt * 16 + g * 8 + r;
      float v = acc[rt][r] * invn[row];
      if (FIRST) v = fmaxf(v, 0.0f);          // ReLU before BatchNorm
      out[(size_t)(m0 + row) * NH + col] = v;
      if (FIRST) { s1 += v; s2 += v * v; }
    }
  }
  if (FIRST) {
    s1 += __shfl_xor(s1, 16, 32);             // merge row-halves (same column)
    s2 += __shfl_xor(s2, 16, 32);
    if (g == 0) {
      atomicAdd(&bnsum[col], s1);
      atomicAdd(&bnsq[col], s2);
    }
  }
}

// ---------------------------------------------------------------------------
extern "C" void kernel_launch(void* const* d_in, const int* in_sizes, int n_in,
                              void* d_out, int out_size, void* d_ws, size_t ws_size,
                              hipStream_t stream) {
  const float*     x     = (const float*)d_in[0];
  const long long* ei    = (const long long*)d_in[1];   // int64 [2, E]
  const float*     W1l   = (const float*)d_in[2];
  const float*     b1    = (const float*)d_in[3];
  const float*     W1r   = (const float*)d_in[4];
  const float*     gamma = (const float*)d_in[5];
  const float*     beta  = (const float*)d_in[6];
  const float*     W2l   = (const float*)d_in[7];
  const float*     b2    = (const float*)d_in[8];
  const float*     W2r   = (const float*)d_in[9];

  float* ws    = (float*)d_ws;
  float* deg   = ws + OFF_DEG;
  float* bnsum = ws + OFF_BNSUM;
  float* bnsq  = ws + OFF_BNSQ;
  float* agg1  = ws + OFF_AGG1;
  float* agg2  = ws + OFF_AGG2;
  float* h     = ws + OFF_H;
  __bf16* W1l_bf = (__bf16*)(ws + F_TOT);
  __bf16* W1r_bf = W1l_bf + NF * NH;
  __bf16* W2l_bf = W1r_bf + NF * NH;
  __bf16* W2r_bf = W2l_bf + NH * NH;

  // 0) zero accumulators; swizzle weights into B-fragment bf16 layout
  k_zero<<<(unsigned)((ZERO_TOT + 255) / 256), 256, 0, stream>>>(ws, ZERO_TOT);
  k_swizzleW<NF><<<(NF * NH) / 256, 256, 0, stream>>>(W1l, W1l_bf);
  k_swizzleW<NF><<<(NF * NH) / 256, 256, 0, stream>>>(W1r, W1r_bf);
  k_swizzleW<NH><<<(NH * NH) / 256, 256, 0, stream>>>(W2l, W2l_bf);
  k_swizzleW<NH><<<(NH * NH) / 256, 256, 0, stream>>>(W2r, W2r_bf);

  // 1) degree + neighbor-sum scatter (layer 1)
  k_degree<<<(Ee + 255) / 256, 256, 0, stream>>>(ei, deg);
  k_scatter128<<<(Ee * 32) / 256, 256, 0, stream>>>(ei, x, agg1);

  // 2) SAGE layer 1: GEMM pair + L2 norm + ReLU + BN stats (WMMA bf16)
  k_sage_gemm<NF, true><<<Nn / ROWS, 512, 0, stream>>>(agg1, x, deg, W1l_bf, W1r_bf,
                                                       b1, h, bnsum, bnsq);

  // 3) BatchNorm apply (in place on h)
  k_bn_apply<<<((size_t)Nn * NH) / 256, 256, 0, stream>>>(h, bnsum, bnsq, gamma, beta);

  // 4) neighbor-sum scatter (layer 2, 256 feats)
  k_scatter256<<<(Ee * 32) / 256, 256, 0, stream>>>(ei, h, agg2);

  // 5) SAGE layer 2: GEMM pair + L2 norm -> d_out (WMMA bf16)
  k_sage_gemm<NH, false><<<Nn / ROWS, 512, 0, stream>>>(agg2, h, deg, W2l_bf, W2r_bf,
                                                        b2, (float*)d_out, bnsum, bnsq);
}
